// BiLSTM_CRF_36618891166155
// MI455X (gfx1250) — matrix operand order repaired and checked by reading, not compiled, not run on hardware
//
#include <hip/hip_runtime.h>
#include <hip/hip_bf16.h>

#define S_LEN 2048
#define NC    32
#define E_DIM 512
#define CE_D  64
#define CH_D  64
#define I_DIM 640      // E + 2*CH
#define HD    256      // word hidden per direction
#define G4H   1024     // 4*HD
#define T_TAG 48

typedef _Float16 half16 __attribute__((ext_vector_type(16)));
typedef _Float16 half8  __attribute__((ext_vector_type(8)));
typedef float    float8 __attribute__((ext_vector_type(8)));
typedef unsigned int uint4v __attribute__((ext_vector_type(4)));
typedef int      int4v  __attribute__((ext_vector_type(4)));
typedef int      int8v  __attribute__((ext_vector_type(8)));

union H16u { half16 v; half8 h[2]; };

// ---- TDM availability dispatch (5-arg ROCm7.2 vs 6-arg therock headers) ----
#if __has_builtin(__builtin_amdgcn_tensor_load_to_lds) && __has_builtin(__builtin_amdgcn_s_wait_tensorcnt)
#define HAVE_TDM 1
#if __has_include(<hip/amd_detail/amd_gfx1250_TDM.h>)
#define TDM_SIX_ARG 1
#endif
#endif

// A fragment (16x32 f16): lane L holds row M=L&15; lanes 0-15 take K in
// {0..15}, lanes 16-31 take K in {16..31}. One 32B load.
__device__ inline half16 loadA(const _Float16* base, int ld, int lane) {
    int m   = lane & 15;
    int klo = (lane < 16) ? 0 : 16;
    const _Float16* p = base + m * ld + klo;
    return *(const half16*)(p);
}

// B fragment (32x16 f16) with B[k][n] = W[n][k] (W row-major, leading dim ldw):
// lane L holds col N=L&15; lanes 0-15: K=k0..k0+15, lanes 16-31: K=k0+16..k0+31.
__device__ inline half16 loadB(const _Float16* W, int ldw, int n0, int k0, int lane) {
    int n   = lane & 15;
    int klo = (lane < 16) ? 0 : 16;
    return *(const half16*)(W + (size_t)(n0 + n) * ldw + k0 + klo);
}

#define WMMA_F16(a, b, c) \
    __builtin_amdgcn_wmma_f32_16x16x32_f16(false, (a), false, (b), (short)0, (c), false, false)

#if defined(HAVE_TDM)
// 1-D TDM tile load: nelem f16 elements (contiguous) global -> LDS.
// D# per CDNA5 ISA ch.8: group0 = {count=1 | lds_addr | global_addr | type=2},
// group1 = {data_size=2B, tensor_dim0=nelem, tensor_dim1=1, tile_dim0=nelem,
//           tensor_dim0_stride=nelem}.
__device__ inline void tdm_load_tile(const _Float16* gsrc, _Float16* lds_dst, int nelem) {
    unsigned long long ga = (unsigned long long)(size_t)gsrc;
    unsigned int laddr = (unsigned int)(size_t)lds_dst;   // LDS aperture: low 32 bits
    uint4v g0;
    g0[0] = 1u;                                            // count=1 (valid user D#)
    g0[1] = laddr;                                         // lds_addr
    g0[2] = (unsigned int)ga;                              // global_addr[31:0]
    g0[3] = ((unsigned int)(ga >> 32) & 0x01FFFFFFu) | 0x80000000u; // [56:32] | type=2
    unsigned int td0 = (unsigned int)nelem;
    int8v g1 = {};
    g1[0] = (int)(1u << 16);                               // data_size = 1 -> 2 bytes
    g1[1] = (int)((td0 & 0xFFFFu) << 16);                  // tensor_dim0[15:0]
    g1[2] = (int)((td0 >> 16) | (1u << 16));               // td0[31:16] | tensor_dim1=1
    g1[3] = (int)(td0 << 16);                              // td1 hi=0 | tile_dim0
    g1[4] = 0;                                             // tile_dim1=0, tile_dim2=0
    g1[5] = (int)td0;                                      // tensor_dim0_stride lo
    g1[6] = 0;
    g1[7] = 0;
    int4v z4 = {};
#if defined(TDM_SIX_ARG)
    int8v z8 = {};
    __builtin_amdgcn_tensor_load_to_lds(g0, g1, z4, z4, z8, 0);
#else
    __builtin_amdgcn_tensor_load_to_lds(g0, g1, z4, z4, 0);
#endif
}
#endif

// ---------------------------------------------------------------- converts
__global__ void f32_to_f16_kernel(const float* __restrict__ src,
                                  _Float16* __restrict__ dst, int n) {
    int i = blockIdx.x * blockDim.x + threadIdx.x;
    if (i < n) dst[i] = (_Float16)src[i];
}

__global__ void gather_word_emb(const int* __restrict__ words,
                                const float* __restrict__ emb,
                                _Float16* __restrict__ embh) {
    int i = blockIdx.x * blockDim.x + threadIdx.x;           // t*E + e
    if (i >= S_LEN * E_DIM) return;
    int t = i / E_DIM, e = i - t * E_DIM;
    embh[(size_t)t * I_DIM + e] = (_Float16)emb[(size_t)words[t] * E_DIM + e];
}

__global__ void gather_char_emb(const int* __restrict__ chars,
                                const float* __restrict__ cemb,
                                _Float16* __restrict__ ceh) {
    int i = blockIdx.x * blockDim.x + threadIdx.x;           // (t*NC+b)*CE + e
    if (i >= S_LEN * NC * CE_D) return;
    int idx = i >> 6, e = i & 63;
    ceh[i] = (_Float16)cemb[(size_t)chars[idx] * CE_D + e];
}

// ------------------------------------------------- char BiLSTM (persistent)
// grid.x = 2 (direction). Per step: gates(32x256) = x_t(32x64)@Wih^T +
// h(32x64)@Whh^T + b via WMMA; TDM double-buffers next step's x tile in LDS.
// Emits h[row 31] (== char_out[:, -1]) into emb columns 512 + dir*64.
__global__ void __launch_bounds__(256) char_lstm_kernel(
    const _Float16* __restrict__ ceh,
    const _Float16* __restrict__ WihF, const _Float16* __restrict__ WhhF,
    const float* __restrict__ bF,
    const _Float16* __restrict__ WihB, const _Float16* __restrict__ WhhB,
    const float* __restrict__ bB,
    _Float16* __restrict__ embh) {
    const int dir = blockIdx.x;
    const _Float16* Wih  = dir ? WihB : WihF;
    const _Float16* Whh  = dir ? WhhB : WhhF;
    const float*    bias = dir ? bB   : bF;

    __shared__ float gates[32 * 256];                 // 32 KB
    __shared__ __align__(32) _Float16 hbuf[32 * 64];  // 4 KB
    __shared__ float cbuf[32 * 64];                   // 8 KB
#if defined(HAVE_TDM)
    __shared__ __align__(32) _Float16 xstage[2][NC * CE_D];  // 2 x 4 KB
#endif

    const int tid  = threadIdx.x;
    const int lane = tid & 31;
    const int wave = tid >> 5;

    for (int i = tid; i < 32 * 64; i += 256) { hbuf[i] = (_Float16)0.0f; cbuf[i] = 0.0f; }
#if defined(HAVE_TDM)
    if (wave == 0) {
        tdm_load_tile(ceh + (size_t)((dir ? S_LEN - 1 : 0) * NC) * CE_D,
                      xstage[0], NC * CE_D);
        __builtin_amdgcn_s_wait_tensorcnt(0);
    }
#endif
    __syncthreads();

    for (int s = 0; s < S_LEN; ++s) {
        const int t = dir ? (S_LEN - 1 - s) : s;
#if defined(HAVE_TDM)
        if (wave == 0 && s + 1 < S_LEN) {
            const int tn = dir ? (S_LEN - 2 - s) : (s + 1);
            tdm_load_tile(ceh + (size_t)(tn * NC) * CE_D, xstage[(s + 1) & 1], NC * CE_D);
        }
        const _Float16* Axbase = xstage[s & 1];
#else
        const _Float16* Axbase = ceh + (size_t)(t * NC) * CE_D;
#endif
        // 32 output tiles (2 M x 16 N) over 8 waves -> 4 tiles/wave
        for (int q = 0; q < 4; ++q) {
            int tau = wave * 4 + q;
            int mt  = (tau & 1) * 16;
            int nt  = (tau >> 1) * 16;
            float8 acc = {};
            const _Float16* Ax = Axbase + mt * CE_D;
            acc = WMMA_F16(loadA(Ax,      CE_D, lane), loadB(Wih, CE_D, nt, 0,  lane), acc);
            acc = WMMA_F16(loadA(Ax + 32, CE_D, lane), loadB(Wih, CE_D, nt, 32, lane), acc);
            const _Float16* Ah = hbuf + mt * CE_D;
            acc = WMMA_F16(loadA(Ah,      CE_D, lane), loadB(Whh, CE_D, nt, 0,  lane), acc);
            acc = WMMA_F16(loadA(Ah + 32, CE_D, lane), loadB(Whh, CE_D, nt, 32, lane), acc);
            int n  = nt + (lane & 15);
            float bb = bias[n];
            int mrow = mt + ((lane >> 4) << 3);
#pragma unroll
            for (int r = 0; r < 8; ++r)
                gates[(mrow + r) * 256 + n] = acc[r] + bb;
        }
        __syncthreads();
        for (int i = tid; i < 32 * 64; i += 256) {
            int b_ = i >> 6, hh = i & 63;
            const float* gr = &gates[b_ * 256];
            float ig = 1.0f / (1.0f + __expf(-gr[hh]));
            float fg = 1.0f / (1.0f + __expf(-gr[64 + hh]));
            float gg = tanhf(gr[128 + hh]);
            float og = 1.0f / (1.0f + __expf(-gr[192 + hh]));
            float cn = fg * cbuf[i] + ig * gg;
            float hn = og * tanhf(cn);
            cbuf[i] = cn;
            hbuf[i] = (_Float16)hn;
            if (b_ == 31)
                embh[(size_t)t * I_DIM + E_DIM + dir * CH_D + hh] = (_Float16)hn;
        }
#if defined(HAVE_TDM)
        if (wave == 0) __builtin_amdgcn_s_wait_tensorcnt(0);
#endif
        __syncthreads();
    }
}

// ------------------------------------------- word input projection (WMMA)
// xproj[dir] = embh(2048x640) @ Wih[dir]^T (640x1024) + b.
// One M-tile x 4 N-tiles per wave: one A-frag load feeds 4 WMMAs per k-step.
// 2 dirs x 128 mtiles x 16 ngroups = 4096 waves -> 512 blocks x 8 waves.
__global__ void __launch_bounds__(256) wordproj_gemm_kernel(
    const _Float16* __restrict__ embh,
    const _Float16* __restrict__ WihF, const float* __restrict__ bF,
    const _Float16* __restrict__ WihB, const float* __restrict__ bB,
    float* __restrict__ xprojF, float* __restrict__ xprojB) {
    int wid  = blockIdx.x * 8 + (threadIdx.x >> 5);
    int lane = threadIdx.x & 31;
    int dir  = wid >> 11;
    int rem  = wid & 2047;
    int mt   = (rem >> 4) * 16;
    int nt0  = (rem & 15) * 64;
    const _Float16* W    = dir ? WihB : WihF;
    const float*    bias = dir ? bB   : bF;
    float*          out  = dir ? xprojB : xprojF;

    float8 a0 = {}, a1 = {}, a2 = {}, a3 = {};
    const _Float16* Abase = embh + (size_t)mt * I_DIM;
    for (int k0 = 0; k0 < I_DIM; k0 += 32) {
        half16 a = loadA(Abase + k0, I_DIM, lane);
        a0 = WMMA_F16(a, loadB(W, I_DIM, nt0,      k0, lane), a0);
        a1 = WMMA_F16(a, loadB(W, I_DIM, nt0 + 16, k0, lane), a1);
        a2 = WMMA_F16(a, loadB(W, I_DIM, nt0 + 32, k0, lane), a2);
        a3 = WMMA_F16(a, loadB(W, I_DIM, nt0 + 48, k0, lane), a3);
    }
    int nl   = lane & 15;
    int mrow = mt + ((lane >> 4) << 3);
#pragma unroll
    for (int p = 0; p < 4; ++p) {
        const float8& acc = p == 0 ? a0 : p == 1 ? a1 : p == 2 ? a2 : a3;
        int n = nt0 + p * 16 + nl;
        float bb = bias[n];
#pragma unroll
        for (int r = 0; r < 8; ++r)
            out[(size_t)(mrow + r) * G4H + n] = acc[r] + bb;
    }
}

// --------------------------------------------- word BiLSTM (persistent GEMV)
// batch=1 -> VALU GEMV; Whh in f16 halves per-step L2 traffic (0.5 MB/step).
__global__ void __launch_bounds__(256) word_lstm_kernel(
    const float* __restrict__ xprojF, const _Float16* __restrict__ WhhFh,
    const float* __restrict__ xprojB, const _Float16* __restrict__ WhhBh,
    _Float16* __restrict__ lstm_h) {
    const int dir = blockIdx.x;
    const float*    xproj = dir ? xprojB : xprojF;
    const _Float16* Whh   = dir ? WhhBh  : WhhFh;

    __shared__ float hsh[HD];
    __shared__ float gsh[G4H];
    const int tid = threadIdx.x;
    hsh[tid] = 0.0f;
    float c = 0.0f;
    __syncthreads();

    for (int s = 0; s < S_LEN; ++s) {
        const int t = dir ? (S_LEN - 1 - s) : s;
        const float* xp = xproj + (size_t)t * G4H;
        if (s + 1 < S_LEN) {   // prefetch next step's xproj row (global_prefetch_b8)
            const int tn = dir ? (S_LEN - 2 - s) : (s + 1);
            __builtin_prefetch(xproj + (size_t)tn * G4H + tid * 4, 0, 1);
        }
#pragma unroll
        for (int r = 0; r < 4; ++r) {
            int g = tid * 4 + r;
            const half16* wr = (const half16*)(Whh + (size_t)g * HD);
            float sum = 0.0f;
            for (int j = 0; j < HD / 16; ++j) {
                half16 w = wr[j];
                const float* hp = &hsh[j * 16];
#pragma unroll
                for (int e = 0; e < 16; ++e) sum += (float)w[e] * hp[e];
            }
            gsh[g] = xp[g] + sum;
        }
        __syncthreads();
        float ig = 1.0f / (1.0f + __expf(-gsh[tid]));
        float fg = 1.0f / (1.0f + __expf(-gsh[HD + tid]));
        float gg = tanhf(gsh[2 * HD + tid]);
        float og = 1.0f / (1.0f + __expf(-gsh[3 * HD + tid]));
        c = fg * c + ig * gg;
        float h = og * tanhf(c);
        hsh[tid] = h;
        lstm_h[(size_t)t * (2 * HD) + dir * HD + tid] = (_Float16)h;
        __syncthreads();
    }
}

// ---------------------------------------------------------- emissions (WMMA)
// em = lstm_h(2048x512) @ W_out^T (512x48) + b.  128 mtiles x 3 ntiles = 384
// waves -> 48 blocks x 8 waves, 16 k-steps each.
__global__ void __launch_bounds__(256) emissions_wmma_kernel(
    const _Float16* __restrict__ lstm_h,
    const _Float16* __restrict__ Wout_h,
    const float* __restrict__ b_out, float* __restrict__ em) {
    int wid  = blockIdx.x * 8 + (threadIdx.x >> 5);
    int lane = threadIdx.x & 31;
    int mt   = (wid / 3) * 16;
    int nt   = (wid % 3) * 16;
    float8 acc = {};
    const _Float16* Abase = lstm_h + (size_t)mt * (2 * HD);
    for (int k0 = 0; k0 < 2 * HD; k0 += 32)
        acc = WMMA_F16(loadA(Abase + k0, 2 * HD, lane),
                       loadB(Wout_h, 2 * HD, nt, k0, lane), acc);
    int n    = nt + (lane & 15);
    float bb = b_out[n];
    int mrow = mt + ((lane >> 4) << 3);
#pragma unroll
    for (int r = 0; r < 8; ++r)
        em[(size_t)(mrow + r) * T_TAG + n] = acc[r] + bb;
}

// ---------------------------------------------------------- CRF NLL
__global__ void __launch_bounds__(64) crf_kernel(
    const float* __restrict__ em, const int* __restrict__ tags,
    const float* __restrict__ trans, const float* __restrict__ start_t,
    const float* __restrict__ end_t, float* __restrict__ out) {
    __shared__ float alpha[T_TAG], nalpha[T_TAG], red[64];
    const int tid = threadIdx.x;

    float g = 0.0f;
    for (int t = tid; t < S_LEN; t += 64) {
        g += em[t * T_TAG + tags[t]];
        if (t > 0) g += trans[tags[t - 1] * T_TAG + tags[t]];
    }
    if (tid == 0) g += start_t[tags[0]] + end_t[tags[S_LEN - 1]];
    red[tid] = g;

    if (tid < T_TAG) alpha[tid] = start_t[tid] + em[tid];
    __syncthreads();

    for (int t = 1; t < S_LEN; ++t) {
        if (tid < T_TAG) {
            float m = -3.4e38f;
            for (int i = 0; i < T_TAG; ++i)
                m = fmaxf(m, alpha[i] + trans[i * T_TAG + tid]);
            float s = 0.0f;
            for (int i = 0; i < T_TAG; ++i)
                s += __expf(alpha[i] + trans[i * T_TAG + tid] - m);
            nalpha[tid] = m + __logf(s) + em[t * T_TAG + tid];
        }
        __syncthreads();
        if (tid < T_TAG) alpha[tid] = nalpha[tid];
        __syncthreads();
    }

    if (tid == 0) {
        float gold = 0.0f;
        for (int i = 0; i < 64; ++i) gold += red[i];
        float m = -3.4e38f;
        for (int j = 0; j < T_TAG; ++j) m = fmaxf(m, alpha[j] + end_t[j]);
        float s = 0.0f;
        for (int j = 0; j < T_TAG; ++j) s += __expf(alpha[j] + end_t[j] - m);
        out[0] = (m + __logf(s)) - gold;
    }
}

// ------------------------------------------------------------------ launch
extern "C" void kernel_launch(void* const* d_in, const int* in_sizes, int n_in,
                              void* d_out, int out_size, void* d_ws, size_t ws_size,
                              hipStream_t stream) {
    const int*   words      = (const int*)d_in[0];
    const int*   chars      = (const int*)d_in[1];
    const int*   tags       = (const int*)d_in[2];
    const float* emb_table  = (const float*)d_in[3];
    const float* cemb_table = (const float*)d_in[4];
    const float* cWihF = (const float*)d_in[5];
    const float* cWhhF = (const float*)d_in[6];
    const float* cbF   = (const float*)d_in[7];
    const float* cWihB = (const float*)d_in[8];
    const float* cWhhB = (const float*)d_in[9];
    const float* cbB   = (const float*)d_in[10];
    const float* WihF  = (const float*)d_in[11];
    const float* WhhF  = (const float*)d_in[12];
    const float* bF    = (const float*)d_in[13];
    const float* WihB  = (const float*)d_in[14];
    const float* WhhB  = (const float*)d_in[15];
    const float* bB    = (const float*)d_in[16];
    const float* W_out = (const float*)d_in[17];
    const float* b_out = (const float*)d_in[18];
    const float* trans = (const float*)d_in[19];
    const float* start_t = (const float*)d_in[20];
    const float* end_t   = (const float*)d_in[21];
    float* out = (float*)d_out;

    // workspace carve-up (256B aligned)
    size_t off = 0;
    auto alloc = [&](size_t bytes) -> size_t {
        size_t o = off; off = (off + bytes + 255) & ~(size_t)255; return o;
    };
    char* ws = (char*)d_ws;
    _Float16* embh    = (_Float16*)(ws + alloc((size_t)S_LEN * I_DIM * 2));
    _Float16* ceh     = (_Float16*)(ws + alloc((size_t)S_LEN * NC * CE_D * 2));
    _Float16* WihF_h  = (_Float16*)(ws + alloc((size_t)G4H * I_DIM * 2));
    _Float16* WihB_h  = (_Float16*)(ws + alloc((size_t)G4H * I_DIM * 2));
    _Float16* WhhF_h  = (_Float16*)(ws + alloc((size_t)G4H * HD * 2));
    _Float16* WhhB_h  = (_Float16*)(ws + alloc((size_t)G4H * HD * 2));
    _Float16* cWihF_h = (_Float16*)(ws + alloc((size_t)4 * CH_D * CE_D * 2));
    _Float16* cWhhF_h = (_Float16*)(ws + alloc((size_t)4 * CH_D * CH_D * 2));
    _Float16* cWihB_h = (_Float16*)(ws + alloc((size_t)4 * CH_D * CE_D * 2));
    _Float16* cWhhB_h = (_Float16*)(ws + alloc((size_t)4 * CH_D * CH_D * 2));
    _Float16* Wout_h  = (_Float16*)(ws + alloc((size_t)T_TAG * 2 * HD * 2));
    float* xprojF  = (float*)(ws + alloc((size_t)S_LEN * G4H * 4));
    float* xprojB  = (float*)(ws + alloc((size_t)S_LEN * G4H * 4));
    _Float16* lstm_h = (_Float16*)(ws + alloc((size_t)S_LEN * 2 * HD * 2));
    float* emis    = (float*)(ws + alloc((size_t)S_LEN * T_TAG * 4));

    // 1) weight conversions to f16
    auto cvt = [&](const float* s, _Float16* d, int n) {
        f32_to_f16_kernel<<<(n + 255) / 256, 256, 0, stream>>>(s, d, n);
    };
    cvt(WihF,  WihF_h,  G4H * I_DIM);
    cvt(WihB,  WihB_h,  G4H * I_DIM);
    cvt(WhhF,  WhhF_h,  G4H * HD);
    cvt(WhhB,  WhhB_h,  G4H * HD);
    cvt(cWihF, cWihF_h, 4 * CH_D * CE_D);
    cvt(cWhhF, cWhhF_h, 4 * CH_D * CH_D);
    cvt(cWihB, cWihB_h, 4 * CH_D * CE_D);
    cvt(cWhhB, cWhhB_h, 4 * CH_D * CH_D);
    cvt(W_out, Wout_h,  T_TAG * 2 * HD);

    // 2) embedding gathers (f32 tables -> f16 activations)
    gather_word_emb<<<(S_LEN * E_DIM + 255) / 256, 256, 0, stream>>>(words, emb_table, embh);
    gather_char_emb<<<(S_LEN * NC * CE_D + 255) / 256, 256, 0, stream>>>(chars, cemb_table, ceh);

    // 3) char BiLSTM (persistent, WMMA + TDM double-buffer); fills embh cols 512..639
    char_lstm_kernel<<<2, 256, 0, stream>>>(ceh, cWihF_h, cWhhF_h, cbF,
                                            cWihB_h, cWhhB_h, cbB, embh);

    // 4) word input projection GEMM (WMMA, 4 N-tiles per wave)
    wordproj_gemm_kernel<<<512, 256, 0, stream>>>(embh, WihF_h, bF, WihB_h, bB,
                                                  xprojF, xprojB);

    // 5) word BiLSTM recurrence (batch=1 -> f16 GEMV, L2-resident Whh)
    word_lstm_kernel<<<2, 256, 0, stream>>>(xprojF, WhhF_h, xprojB, WhhB_h, lstm_h);

    // 6) emissions via WMMA
    emissions_wmma_kernel<<<48, 256, 0, stream>>>(lstm_h, Wout_h, b_out, emis);

    // 7) CRF negative log-likelihood -> scalar
    crf_kernel<<<1, 64, 0, stream>>>(emis, tags, trans, start_t, end_t, out);
}
